// FusedGWLoss_10866267258903
// MI455X (gfx1250) — matrix operand order repaired and checked by reading, not compiled, not run on hardware
//
#include <hip/hip_runtime.h>
#include <hip/hip_bf16.h>

#define N1 2048
#define N2 2048
#define DD 256

typedef __attribute__((ext_vector_type(16))) __bf16 v16bf;
typedef __attribute__((ext_vector_type(8)))  float  v8f;

#define OUT_F32    0
#define OUT_EXPNEG 1
#define OUT_BF16_T 2

// ---------------- WMMA fragment loads (CDNA5 wave32 layouts, CDNA5 ISA) ----
// 16-bit A 16x32: lane l(0..15) row M=l holds K {0..7,16..23}; lane l+16 holds
// K {8..15,24..31}.  Two contiguous 16B loads per lane.
__device__ __forceinline__ v16bf load_a_frag(const __bf16* A, int lda,
                                             int row0, int k0, int lane) {
  const int r  = lane & 15;
  const int hi = lane >> 4;
  const __bf16* p = A + (size_t)(row0 + r) * lda + k0 + 8 * hi;
  union { uint4 q[2]; v16bf v; } u;
  u.q[0] = *reinterpret_cast<const uint4*>(p);
  u.q[1] = *reinterpret_cast<const uint4*>(p + 16);
  return u.v;
}
// 16-bit B 32x16: lane l(0..15) column N=l holds K {0..7,16..23}; lane l+16 holds
// K {8..15,24..31}.  B column j == row j of the row-major source (A*B^T formulation),
// so this is two contiguous 16B loads per lane.
__device__ __forceinline__ v16bf load_b_frag(const __bf16* B, int ldb,
                                             int col0, int k0, int lane) {
  const int r  = lane & 15;
  const int hi = lane >> 4;
  const __bf16* p = B + (size_t)(col0 + r) * ldb + k0 + 8 * hi;
  union { uint4 q[2]; v16bf v; } u;
  u.q[0] = *reinterpret_cast<const uint4*>(p);
  u.q[1] = *reinterpret_cast<const uint4*>(p + 16);
  return u.v;
}

// ---------------- C = A(MxK) * B(NxK)^T, bf16 in, f32 accumulate -----------
// 256 threads = 8 waves; waves tiled 2(M)x4(N); each wave owns a 64x32 tile
// (4x2 WMMA tiles) -> WG tile 128x128.  Register double-buffered over K
// (stride 64 = two 32-wide halves) so loads for the next half are in flight
// while the current half's 8 WMMAs issue.
__global__ __launch_bounds__(256)
void gemm_abt_wmma(const __bf16* __restrict__ A, const __bf16* __restrict__ B,
                   int Kdim, int mode, float* __restrict__ outF,
                   __bf16* __restrict__ outB, int ldc) {
  const int lane = threadIdx.x & 31;
  const int wave = threadIdx.x >> 5;
  const int wm = wave & 1;                  // 0..1
  const int wn = wave >> 1;                 // 0..3
  const int rowW = blockIdx.x * 128 + wm * 64;
  const int colW = blockIdx.y * 128 + wn * 32;

  v8f acc[4][2] = {};
  v16bf afE[4], bfE[2], afO[4], bfO[2];

  // preload even half k=0
#pragma unroll
  for (int mi = 0; mi < 4; ++mi)
    afE[mi] = load_a_frag(A, Kdim, rowW + mi * 16, 0, lane);
  bfE[0] = load_b_frag(B, Kdim, colW,      0, lane);
  bfE[1] = load_b_frag(B, Kdim, colW + 16, 0, lane);

  for (int k0 = 0; k0 < Kdim; k0 += 64) {
    // issue loads for odd half (k0+32) before consuming even half
#pragma unroll
    for (int mi = 0; mi < 4; ++mi)
      afO[mi] = load_a_frag(A, Kdim, rowW + mi * 16, k0 + 32, lane);
    bfO[0] = load_b_frag(B, Kdim, colW,      k0 + 32, lane);
    bfO[1] = load_b_frag(B, Kdim, colW + 16, k0 + 32, lane);
    if (k0 + 64 < Kdim) {                   // global_prefetch_b8 next K slab
      __builtin_prefetch(A + (size_t)(rowW + (lane & 15)) * Kdim + k0 + 64, 0, 1);
      __builtin_prefetch(B + (size_t)(colW + (lane & 15)) * Kdim + k0 + 64, 0, 1);
    }
    // compute even half
#pragma unroll
    for (int mi = 0; mi < 4; ++mi)
#pragma unroll
      for (int ni = 0; ni < 2; ++ni)
        acc[mi][ni] = __builtin_amdgcn_wmma_f32_16x16x32_bf16(
            false, afE[mi], false, bfE[ni], (short)0, acc[mi][ni], false, false);
    // issue loads for next even half (k0+64) before consuming odd half
    if (k0 + 64 < Kdim) {
#pragma unroll
      for (int mi = 0; mi < 4; ++mi)
        afE[mi] = load_a_frag(A, Kdim, rowW + mi * 16, k0 + 64, lane);
      bfE[0] = load_b_frag(B, Kdim, colW,      k0 + 64, lane);
      bfE[1] = load_b_frag(B, Kdim, colW + 16, k0 + 64, lane);
    }
    // compute odd half
#pragma unroll
    for (int mi = 0; mi < 4; ++mi)
#pragma unroll
      for (int ni = 0; ni < 2; ++ni)
        acc[mi][ni] = __builtin_amdgcn_wmma_f32_16x16x32_bf16(
            false, afO[mi], false, bfO[ni], (short)0, acc[mi][ni], false, false);
  }

  // C/D layout: VGPR vr, lanes 0-15 -> (M=vr, N=lane); lanes 16-31 -> (M=vr+8)
  const int r  = lane & 15;
  const int hi = lane >> 4;
#pragma unroll
  for (int mi = 0; mi < 4; ++mi) {
#pragma unroll
    for (int ni = 0; ni < 2; ++ni) {
      const int rb = rowW + mi * 16;
      const int cb = colW + ni * 16;
#pragma unroll
      for (int vr = 0; vr < 8; ++vr) {
        const int m = rb + vr + 8 * hi;
        const int n = cb + r;
        const float cv = acc[mi][ni][vr];
        if (mode == OUT_F32)         outF[(size_t)m * ldc + n] = cv;
        else if (mode == OUT_EXPNEG) outF[(size_t)m * ldc + n] = __expf(-cv);
        else                         outB[(size_t)n * ldc + m] = (__bf16)cv;
      }
    }
  }
}

// ---------------- small helper kernels -------------------------------------
__global__ void cvt_f32_bf16(const float* __restrict__ in,
                             __bf16* __restrict__ out, int n) {
  int i = blockIdx.x * 256 + threadIdx.x;
  if (i < n) out[i] = (__bf16)in[i];
}

// out[row] = scale * sum_j M[row][j]^2   (symmetric matrices: row==col sums)
__global__ void rowsumsq(const float* __restrict__ M, float* __restrict__ out,
                         int n, float scale) {
  __shared__ float red[256];
  const int row = blockIdx.x;
  float s = 0.f;
  for (int j = threadIdx.x; j < n; j += 256) {
    float x = M[(size_t)row * n + j];
    s += x * x;
  }
  red[threadIdx.x] = s; __syncthreads();
  for (int off = 128; off > 0; off >>= 1) {
    if (threadIdx.x < off) red[threadIdx.x] += red[threadIdx.x + off];
    __syncthreads();
  }
  if (threadIdx.x == 0) out[row] = red[0] * scale;
}

__global__ void init_state(float* __restrict__ s, __bf16* __restrict__ s_bt,
                           float* __restrict__ u, float* __restrict__ v) {
  int i = blockIdx.x * 256 + threadIdx.x;
  const float s0 = 1.0f / ((float)N1 * (float)N2);
  s[i]    = s0;
  s_bt[i] = (__bf16)s0;
  if (i < N1) u[i] = 1.0f / (float)N1;
  if (i < N2) v[i] = 1.0f / (float)N2;
}

// Kb holds M = c1 @ s @ c2^T on entry; overwritten with K; Kt = K^T.
__global__ void kbuild(float* __restrict__ Kb, const float* __restrict__ interC,
                       const float* __restrict__ c1sqa,
                       const float* __restrict__ bc2sq,
                       float* __restrict__ Kt) {
  int idx = blockIdx.x * 256 + threadIdx.x;
  int i = idx >> 11, j = idx & 2047;
  float m    = Kb[idx];
  float L    = c1sqa[i] + bc2sq[j] - 2.0f * m;
  float cost = interC[idx] + 0.01f * L;         // LAMBDA_E = 0.01
  float kv   = __expf(-cost * 20.0f);           // 1/LAMBDA_T = 20
  Kb[idx] = kv;
  Kt[(size_t)j * N1 + i] = kv;
}

// y[row] = num / dot(M[row], x)  — fixed-order tree reduce (deterministic)
__global__ void matvec_recip(const float* __restrict__ M,
                             const float* __restrict__ x,
                             float* __restrict__ y, float num) {
  __shared__ float red[256];
  const int row = blockIdx.x;
  const float* mr = M + (size_t)row * 2048;
  float s = 0.f;
  for (int j = threadIdx.x; j < 2048; j += 256) s += mr[j] * x[j];
  red[threadIdx.x] = s; __syncthreads();
  for (int off = 128; off > 0; off >>= 1) {
    if (threadIdx.x < off) red[threadIdx.x] += red[threadIdx.x + off];
    __syncthreads();
  }
  if (threadIdx.x == 0) y[row] = num / red[0];
}

__global__ void supdate(float* __restrict__ s, const float* __restrict__ Kb,
                        const float* __restrict__ u, const float* __restrict__ v,
                        __bf16* __restrict__ s_bt) {
  int idx = blockIdx.x * 256 + threadIdx.x;
  int i = idx >> 11, j = idx & 2047;
  float sn = 0.05f * s[idx] + 0.95f * u[i] * Kb[idx] * v[j];
  s[idx] = sn;
  s_bt[(size_t)j * N1 + i] = (__bf16)sn;        // keep s^T in bf16 for GEMM1
}

__global__ void loss_partial(const float* __restrict__ interC,
                             const float* __restrict__ s,
                             float* __restrict__ part) {
  __shared__ float red[256];
  const int base = blockIdx.x * 4096;
  float a = 0.f;
  for (int t = threadIdx.x; t < 4096; t += 256) {
    int idx = base + t;
    a += interC[idx] * s[idx];
  }
  red[threadIdx.x] = a; __syncthreads();
  for (int off = 128; off > 0; off >>= 1) {
    if (threadIdx.x < off) red[threadIdx.x] += red[threadIdx.x + off];
    __syncthreads();
  }
  if (threadIdx.x == 0) part[blockIdx.x] = red[0];
}

__global__ void loss_final(const float* __restrict__ part, float* __restrict__ out) {
  __shared__ float red[256];
  float a = 0.f;
  for (int i = threadIdx.x; i < 1024; i += 256) a += part[i];
  red[threadIdx.x] = a; __syncthreads();
  for (int off = 128; off > 0; off >>= 1) {
    if (threadIdx.x < off) red[threadIdx.x] += red[threadIdx.x + off];
    __syncthreads();
  }
  if (threadIdx.x == 0) out[0] = -red[0];
}

// ---------------- host driver ----------------------------------------------
extern "C" void kernel_launch(void* const* d_in, const int* in_sizes, int n_in,
                              void* d_out, int out_size, void* d_ws, size_t ws_size,
                              hipStream_t stream) {
  const float* out1 = (const float*)d_in[0];   // [N1, D]
  const float* out2 = (const float*)d_in[1];   // [N2, D]
  const float* c1   = (const float*)d_in[2];   // [N1, N1] symmetric
  const float* c2   = (const float*)d_in[3];   // [N2, N2] symmetric

  char* ws = (char*)d_ws;
  size_t off = 0;
  auto alloc = [&](size_t bytes) -> void* {
    void* p = ws + off;
    off += (bytes + 255) & ~(size_t)255;
    return p;
  };
  float*  interC = (float*) alloc(sizeof(float)  * N1 * N2);   // exp(-out1 out2^T)
  __bf16* c1b    = (__bf16*)alloc(sizeof(__bf16) * N1 * N1);
  __bf16* c2b    = (__bf16*)alloc(sizeof(__bf16) * N2 * N2);
  float*  sM     = (float*) alloc(sizeof(float)  * N1 * N2);   // transport plan
  __bf16* s_bt   = (__bf16*)alloc(sizeof(__bf16) * N1 * N2);   // s^T bf16
  __bf16* T_b    = (__bf16*)alloc(sizeof(__bf16) * N1 * N2);   // (c1 @ s) bf16
  float*  Kb     = (float*) alloc(sizeof(float)  * N1 * N2);   // M then K
  float*  Kt     = (float*) alloc(sizeof(float)  * N1 * N2);   // K^T
  __bf16* out1b  = (__bf16*)alloc(sizeof(__bf16) * N1 * DD);
  __bf16* out2b  = (__bf16*)alloc(sizeof(__bf16) * N2 * DD);
  float*  u      = (float*) alloc(sizeof(float)  * N1);
  float*  v      = (float*) alloc(sizeof(float)  * N2);
  float*  c1sqa  = (float*) alloc(sizeof(float)  * N1);
  float*  bc2sq  = (float*) alloc(sizeof(float)  * N2);
  float*  part   = (float*) alloc(sizeof(float)  * 1024);

  const dim3 b256(256);
  const int nBig = N1 * N2;                     // 4M elements

  // setup: bf16 copies, squared row sums, state init
  cvt_f32_bf16<<<(N1 * DD) / 256, b256, 0, stream>>>(out1, out1b, N1 * DD);
  cvt_f32_bf16<<<(N2 * DD) / 256, b256, 0, stream>>>(out2, out2b, N2 * DD);
  cvt_f32_bf16<<<nBig / 256, b256, 0, stream>>>(c1, c1b, nBig);
  cvt_f32_bf16<<<nBig / 256, b256, 0, stream>>>(c2, c2b, nBig);
  rowsumsq<<<N1, b256, 0, stream>>>(c1, c1sqa, N1, 1.0f / (float)N1);
  rowsumsq<<<N2, b256, 0, stream>>>(c2, bc2sq, N2, 1.0f / (float)N2);
  init_state<<<nBig / 256, b256, 0, stream>>>(sM, s_bt, u, v);

  const dim3 gemmGrid(N1 / 128, N2 / 128);
  // inter_c = exp(-(out1 @ out2^T))
  gemm_abt_wmma<<<gemmGrid, b256, 0, stream>>>(out1b, out2b, DD, OUT_EXPNEG,
                                               interC, nullptr, N2);

  for (int t = 0; t < 10; ++t) {                // OUT_ITER
    // T^T = s^T @ c1  (c1 symmetric) -> stored transposed => T_b = c1 @ s, bf16
    gemm_abt_wmma<<<gemmGrid, b256, 0, stream>>>(s_bt, c1b, N1, OUT_BF16_T,
                                                 nullptr, T_b, N1);
    // M = T @ c2^T = c1 @ s @ c2^T  (c2 symmetric so also == T @ c2)
    gemm_abt_wmma<<<gemmGrid, b256, 0, stream>>>(T_b, c2b, N2, OUT_F32,
                                                 Kb, nullptr, N2);
    kbuild<<<nBig / 256, b256, 0, stream>>>(Kb, interC, c1sqa, bc2sq, Kt);
    for (int it = 0; it < 5; ++it) {            // IN_ITER
      matvec_recip<<<N1, b256, 0, stream>>>(Kb, v, u, 1.0f / (float)N1);
      matvec_recip<<<N2, b256, 0, stream>>>(Kt, u, v, 1.0f / (float)N2);
    }
    supdate<<<nBig / 256, b256, 0, stream>>>(sM, Kb, u, v, s_bt);
  }

  loss_partial<<<1024, b256, 0, stream>>>(interC, sM, part);
  loss_final<<<1, b256, 0, stream>>>(part, (float*)d_out);
  (void)in_sizes; (void)n_in; (void)out_size; (void)ws_size;
}
